// GridNeRF_17514876634251
// MI455X (gfx1250) — matrix operand order repaired and checked by reading, not compiled, not run on hardware
//
#include <hip/hip_runtime.h>
#include <hip/hip_bf16.h>

typedef _Float16 v16h __attribute__((ext_vector_type(16)));
typedef _Float16 v8h  __attribute__((ext_vector_type(8)));
typedef float    v8f  __attribute__((ext_vector_type(8)));

#define BLOCK_PTS 256

__device__ __forceinline__ int imin(int a, int b) { return a < b ? a : b; }

__device__ __forceinline__ float4 lerp4(float4 a, float4 b, float t) {
    float4 r;
    float u = 1.0f - t;
    r.x = a.x * u + b.x * t;
    r.y = a.y * u + b.y * t;
    r.z = a.z * u + b.z * t;
    r.w = a.w * u + b.w * t;
    return r;
}

// Load one K-chunk (32 K values) of a 16xK f16 A-matrix stored row-major in LDS,
// into the CDNA5 WMMA 16-bit A layout:
//   lanes 0-15  (row m=lane):    halves 0-7 = K[kofs+0..7],  halves 8-15 = K[kofs+16..23]
//   lanes 16-31 (row m=lane-16): halves 0-7 = K[kofs+8..15], halves 8-15 = K[kofs+24..31]
__device__ __forceinline__ v16h load_A_chunk(const _Float16* rows, int ldk, int kofs, int lane) {
    int m  = lane & 15;
    int kb = kofs + ((lane >> 4) << 3);   // +0 or +8
    const _Float16* p = rows + m * ldk + kb;
    v8h lo = *(const v8h*)(p);
    v8h hi = *(const v8h*)(p + 16);
    v16h a;
#pragma unroll
    for (int i = 0; i < 8; ++i) { a[i] = lo[i]; a[i + 8] = hi[i]; }
    return a;
}

// Load one K-chunk (32 K values) of a B-matrix column (output feature n) from
// weights stored transposed (out-feature-major) in LDS: WT[n][0..K-1].
//   lanes 0-15  (col n=lane):    halves 0-7 = K[kofs+0..7],  halves 8-15 = K[kofs+16..23]
//   lanes 16-31 (col n=lane-16): halves 0-7 = K[kofs+8..15], halves 8-15 = K[kofs+24..31]
__device__ __forceinline__ v16h load_B_chunk(const _Float16* wt, int K, int n, int kofs, int lane) {
    int kb = kofs + ((lane >> 4) << 3);   // +0 or +8
    const _Float16* p = wt + n * K + kb;
    v8h lo = *(const v8h*)(p);
    v8h hi = *(const v8h*)(p + 16);
    v16h b;
#pragma unroll
    for (int i = 0; i < 8; ++i) { b[i] = lo[i]; b[i + 8] = hi[i]; }
    return b;
}

__device__ __forceinline__ v8f wmma_f16(v16h a, v16h b, v8f c) {
    return __builtin_amdgcn_wmma_f32_16x16x32_f16(false, a, false, b, (short)0, c, false, false);
}

__global__ __launch_bounds__(BLOCK_PTS)
void gridnerf_fused_kernel(const float* __restrict__ coords,
                           const float* __restrict__ g0, const float* __restrict__ g1,
                           const float* __restrict__ g2, const float* __restrict__ g3,
                           const float* __restrict__ g4, const float* __restrict__ g5,
                           const float* __restrict__ g6, const float* __restrict__ g7,
                           const float* __restrict__ W0, const float* __restrict__ b0,
                           const float* __restrict__ W1, const float* __restrict__ b1,
                           const float* __restrict__ W2, const float* __restrict__ b2,
                           const float* __restrict__ W3, const float* __restrict__ b3,
                           float* __restrict__ rgb_out, float* __restrict__ dens_out,
                           float* __restrict__ feats_out, int N)
{
    // --- LDS ---
    __shared__ alignas(16) _Float16 sFeat[BLOCK_PTS * 32];   // f16 features, row-major per point
    __shared__ alignas(16) _Float16 sWT0[64 * 32];           // W0^T  [out=64][in=32]
    __shared__ alignas(16) _Float16 sWT1[64 * 64];           // W1^T
    __shared__ alignas(16) _Float16 sWT2[64 * 64];           // W2^T
    __shared__ alignas(16) float    sW3T[4 * 96];            // W3^T  [out=4][in=96], f32
    __shared__ float sb0[64], sb1[64], sb2[64], sb3[4];
    __shared__ alignas(16) _Float16 sH[8 * 16 * 64];         // per-wave hidden buffer

    const int tid = threadIdx.x;

    // --- Phase 0: cooperative weight preload (transpose + f32->f16) ---
    for (int i = tid; i < 64 * 32; i += BLOCK_PTS) {
        int n = i >> 5, k = i & 31;
        sWT0[i] = (_Float16)W0[k * 64 + n];
    }
    for (int i = tid; i < 64 * 64; i += BLOCK_PTS) {
        int n = i >> 6, k = i & 63;
        sWT1[i] = (_Float16)W1[k * 64 + n];
        sWT2[i] = (_Float16)W2[k * 64 + n];
    }
    for (int i = tid; i < 4 * 96; i += BLOCK_PTS) {
        int o = i / 96, r = i % 96;
        sW3T[i] = W3[r * 4 + o];
    }
    if (tid < 64) { sb0[tid] = b0[tid]; sb1[tid] = b1[tid]; sb2[tid] = b2[tid]; }
    if (tid < 4)  { sb3[tid] = b3[tid]; }

    // --- Phase 1: trilinear sampling, one thread per point ---
    const int p = blockIdx.x * BLOCK_PTS + tid;
    const bool valid = (p < N);
    float cx = 0.f, cy = 0.f, cz = 0.f;
    if (valid) {
        cx = coords[p * 3 + 0];
        cy = coords[p * 3 + 1];
        cz = coords[p * 3 + 2];
    }
    const float* grids[8] = { g0, g1, g2, g3, g4, g5, g6, g7 };
    constexpr int RES[8] = { 16, 32, 64, 128, 128, 128, 128, 128 };

#pragma unroll
    for (int l = 0; l < 8; ++l) {
        const int R = RES[l];
        const float s = (float)(R - 1);
        float gx = fminf(fmaxf(cx * s, 0.f), s);
        float gy = fminf(fmaxf(cy * s, 0.f), s);
        float gz = fminf(fmaxf(cz * s, 0.f), s);
        float fx = floorf(gx), fy = floorf(gy), fz = floorf(gz);
        float wx = gx - fx, wy = gy - fy, wz = gz - fz;
        int x0 = (int)fx, y0 = (int)fy, z0 = (int)fz;
        int x1 = imin(x0 + 1, R - 1), y1 = imin(y0 + 1, R - 1), z1 = imin(z0 + 1, R - 1);
        const float4* G = (const float4*)grids[l];
        float4 c000 = G[(x0 * R + y0) * R + z0];
        float4 c001 = G[(x0 * R + y0) * R + z1];
        float4 c010 = G[(x0 * R + y1) * R + z0];
        float4 c011 = G[(x0 * R + y1) * R + z1];
        float4 c100 = G[(x1 * R + y0) * R + z0];
        float4 c101 = G[(x1 * R + y0) * R + z1];
        float4 c110 = G[(x1 * R + y1) * R + z0];
        float4 c111 = G[(x1 * R + y1) * R + z1];
        float4 c00 = lerp4(c000, c001, wz);
        float4 c01 = lerp4(c010, c011, wz);
        float4 c10 = lerp4(c100, c101, wz);
        float4 c11 = lerp4(c110, c111, wz);
        float4 cc0 = lerp4(c00, c01, wy);
        float4 cc1 = lerp4(c10, c11, wy);
        float4 f   = lerp4(cc0, cc1, wx);

        const int fo = l * 4;
        sFeat[tid * 32 + fo + 0] = (_Float16)f.x;
        sFeat[tid * 32 + fo + 1] = (_Float16)f.y;
        sFeat[tid * 32 + fo + 2] = (_Float16)f.z;
        sFeat[tid * 32 + fo + 3] = (_Float16)f.w;
        if (valid) {
            *(float4*)(feats_out + (size_t)p * 32 + fo) = f;   // feats output (f32)
        }
    }

    __syncthreads();

    // --- Phase 2: WMMA MLP, each wave handles two 16-point tiles ---
    const int wid  = tid >> 5;
    const int lane = tid & 31;
    const int nloc = lane & 15;            // local N within a 16-wide tile (C layout)
    const int m0   = (lane >> 4) << 3;     // C-layout row base (0 or 8)
    _Float16* hw = sH + wid * (16 * 64);   // this wave's private 16x64 h buffer

    for (int it = 0; it < 2; ++it) {
        const int tile = wid * 2 + it;                       // 0..15
        const _Float16* Arows = sFeat + tile * (16 * 32);

        // ---- Layer 0: feats[16x32] @ W0[32x64] -> h[16x64], relu ----
        {
            v16h a0 = load_A_chunk(Arows, 32, 0, lane);
#pragma unroll
            for (int nt = 0; nt < 4; ++nt) {
                int n = nt * 16 + nloc;
                v16h b = load_B_chunk(sWT0, 32, n, 0, lane);
                v8f c = {};
                c = wmma_f16(a0, b, c);
                float bias = sb0[n];
#pragma unroll
                for (int r = 0; r < 8; ++r) {
                    float v = fmaxf(c[r] + bias, 0.f);
                    hw[(m0 + r) * 64 + n] = (_Float16)v;
                }
            }
        }

        // ---- Layers 1 & 2: h[16x64] @ W[64x64] -> h[16x64], relu ----
#pragma unroll
        for (int L = 0; L < 2; ++L) {
            const _Float16* WT = (L == 0) ? sWT1 : sWT2;
            const float*    bb = (L == 0) ? sb1  : sb2;
            // Load full A (both K chunks) before overwriting hw; LDS ops are
            // in-order within a wave, so subsequent stores cannot pass these loads.
            v16h a0 = load_A_chunk(hw, 64, 0, lane);
            v16h a1 = load_A_chunk(hw, 64, 32, lane);
#pragma unroll
            for (int nt = 0; nt < 4; ++nt) {
                int n = nt * 16 + nloc;
                v16h bB0 = load_B_chunk(WT, 64, n, 0, lane);
                v16h bB1 = load_B_chunk(WT, 64, n, 32, lane);
                v8f c = {};
                c = wmma_f16(a0, bB0, c);
                c = wmma_f16(a1, bB1, c);
                float bias = bb[n];
#pragma unroll
                for (int r = 0; r < 8; ++r) {
                    float v = fmaxf(c[r] + bias, 0.f);
                    hw[(m0 + r) * 64 + n] = (_Float16)v;
                }
            }
        }

        // ---- Layer 3: [h(64) | feats(32)] @ W3[96x4] + b3; sigmoid/relu ----
        {
            const int m  = lane & 15;
            const int ob = (lane >> 4) * 2;                  // outputs {0,1} or {2,3}
            const int gp = blockIdx.x * BLOCK_PTS + tile * 16 + m;
            const _Float16* hrow = hw + m * 64;
            const _Float16* frow = sFeat + (tile * 16 + m) * 32;
#pragma unroll
            for (int oo = 0; oo < 2; ++oo) {
                const int o = ob + oo;
                const float* wrow = sW3T + o * 96;
                float acc = sb3[o];
#pragma unroll
                for (int kk = 0; kk < 8; ++kk) {
                    v8h h8 = *(const v8h*)(hrow + kk * 8);
#pragma unroll
                    for (int j = 0; j < 8; ++j) acc += (float)h8[j] * wrow[kk * 8 + j];
                }
#pragma unroll
                for (int kk = 0; kk < 4; ++kk) {
                    v8h f8 = *(const v8h*)(frow + kk * 8);
#pragma unroll
                    for (int j = 0; j < 8; ++j) acc += (float)f8[j] * wrow[64 + kk * 8 + j];
                }
                if (gp < N) {
                    if (o < 3) rgb_out[(size_t)gp * 3 + o] = 1.0f / (1.0f + __expf(-acc));
                    else       dens_out[gp] = fmaxf(acc, 0.f);
                }
            }
        }
    }
}

extern "C" void kernel_launch(void* const* d_in, const int* in_sizes, int n_in,
                              void* d_out, int out_size, void* d_ws, size_t ws_size,
                              hipStream_t stream) {
    const float* coords = (const float*)d_in[0];
    const float* g[8];
    for (int i = 0; i < 8; ++i) g[i] = (const float*)d_in[1 + i];
    const float* W0 = (const float*)d_in[9];
    const float* b0 = (const float*)d_in[10];
    const float* W1 = (const float*)d_in[11];
    const float* b1 = (const float*)d_in[12];
    const float* W2 = (const float*)d_in[13];
    const float* b2 = (const float*)d_in[14];
    const float* W3 = (const float*)d_in[15];
    const float* b3 = (const float*)d_in[16];

    const int N = in_sizes[0] / 3;

    // d_out: rgb [N,3] | density [N] | feats [N,32], all f32, concatenated flat.
    float* out      = (float*)d_out;
    float* rgb      = out;
    float* dens     = out + (size_t)3 * N;
    float* feats    = dens + (size_t)N;

    const int nblocks = (N + BLOCK_PTS - 1) / BLOCK_PTS;
    gridnerf_fused_kernel<<<nblocks, BLOCK_PTS, 0, stream>>>(
        coords, g[0], g[1], g[2], g[3], g[4], g[5], g[6], g[7],
        W0, b0, W1, b1, W2, b2, W3, b3,
        rgb, dens, feats, N);
}
